// Encoder_52046413693112
// MI455X (gfx1250) — compile-verified, matrix operand order checked by reading, not run on hardware
//
#include <hip/hip_runtime.h>

// Problem sizes (fixed by the reference)
#define B_  64
#define T_  512
#define E_  512
#define H_  1024
#define Z_  128
#define G4  4096      // 4*H
#define G2  2048      // 2*H
#define TB  32768     // T*B

typedef __attribute__((ext_vector_type(16))) __bf16 v16bf;
typedef __attribute__((ext_vector_type(8)))  float  v8f;

union Frag16 { uint4 u[2]; v16bf bf; };

// ---- Tensor Data Mover availability ---------------------------------------
#ifndef HAVE_TDM
#if defined(__has_builtin)
#if __has_builtin(__builtin_amdgcn_tensor_load_to_lds)
#define HAVE_TDM 1
#endif
#endif
#endif
#ifndef HAVE_TDM
#define HAVE_TDM 0
#endif

#if HAVE_TDM
typedef unsigned int v4u  __attribute__((ext_vector_type(4)));
typedef int          v8i_ __attribute__((ext_vector_type(8)));
typedef int          v4i_ __attribute__((ext_vector_type(4)));

// Issue one TDM 2-D tile load (bf16 elements): rows x cols from global
// (row stride `stride` elements) into LDS byte offset `lds_off`.
// D# layout per CDNA5 ISA 8.3/8.4: group0 = {flags, lds_addr, global_addr,
// type=2}; group1 = {data_size=2B, tensor_dim0/1, tile_dim0/1, dim0_stride}.
__device__ __forceinline__ void tdm_load_2d_bf16(unsigned lds_off,
                                                 const void* gptr,
                                                 unsigned rows, unsigned cols,
                                                 unsigned stride) {
  const unsigned long long ga = (unsigned long long)(uintptr_t)gptr;
  v4u g0;
  g0[0] = 1u;                                   // count=1 (valid), load, no gather
  g0[1] = lds_off;                              // LDS byte address
  g0[2] = (unsigned)(ga & 0xffffffffu);         // global_addr[31:0]
  g0[3] = (unsigned)((ga >> 32) & 0x01ffffffu)  // global_addr[56:32]
        | (2u << 30);                           // type=2 ("image")
  v8i_ g1;
  g1[0] = (int)(1u << 16);                      // data_size = 1 -> 2 bytes
  g1[1] = (int)(cols << 16);                    // tensor_dim0[15:0] @ [63:48]
  g1[2] = (int)((cols >> 16) | (rows << 16));   // dim0 hi | tensor_dim1 lo
  g1[3] = (int)((rows >> 16) | (cols << 16));   // dim1 hi | tile_dim0
  g1[4] = (int)rows;                            // tile_dim1 (tile_dim2=0 -> 2D)
  g1[5] = (int)stride;                          // tensor_dim0_stride[31:0]
  g1[6] = 0;                                    // stride hi | dim1_stride lo
  g1[7] = 0;
  const v4i_ z4 = {0, 0, 0, 0};
#if __clang_major__ >= 23
  const v8i_ z8 = {0, 0, 0, 0, 0, 0, 0, 0};
  __builtin_amdgcn_tensor_load_to_lds(g0, g1, z4, z4, z8, 0);
#else
  __builtin_amdgcn_tensor_load_to_lds(g0, g1, z4, z4, 0);
#endif
}
#endif  // HAVE_TDM

__device__ __forceinline__ unsigned short f2bf(float f) {
  unsigned int u = __float_as_uint(f);
  u += 0x7fffu + ((u >> 16) & 1u);          // round-to-nearest-even
  return (unsigned short)(u >> 16);
}

__device__ __forceinline__ float sigmoidf_(float x) {
  return 1.0f / (1.0f + __expf(-x));
}

// Load a 16x32 bf16 WMMA operand (A: 16 M-rows x 32 K, or B from N x K
// row-major storage) per the CDNA5 16-bit operand layout:
//   lane = row%16, half = lane>>4
//   v0..3 = K[half*8 + 0..7],  v4..7 = K[16 + half*8 + 0..7]
// -> two 16-byte contiguous loads per lane (global_load_b128 / ds_load_b128).
__device__ __forceinline__ v16bf load_frag(const unsigned short* base, int ld,
                                           int row0, int k0, int lane) {
  const int r    = row0 + (lane & 15);
  const int half = lane >> 4;
  const unsigned short* p = base + (size_t)r * ld + k0 + half * 8;
  Frag16 f;
  f.u[0] = *(const uint4*)(p);
  f.u[1] = *(const uint4*)(p + 16);
  return f.bf;
}

__device__ __forceinline__ v8f wmma_bf16(v16bf a, v16bf b, v8f c) {
  return __builtin_amdgcn_wmma_f32_16x16x32_bf16(
      /*neg_a=*/false, a, /*neg_b=*/false, b,
      /*c_mod=*/(short)0, c, /*reuse_a=*/false, /*reuse_b=*/false);
}

// ---------------------------------------------------------------------------
// Conversion / init kernels
// ---------------------------------------------------------------------------

// (B,T,X) fp32  ->  (T*B, X) bf16   (t-major rows, matching 'tbg' einsum)
__global__ void k_conv_transpose(const float* __restrict__ in,
                                 unsigned short* __restrict__ out, int X) {
  const int t = blockIdx.x, b = blockIdx.y;
  const float* src = in + ((size_t)b * T_ + t) * X;
  unsigned short* dst = out + ((size_t)t * B_ + b) * X;
  for (int x = threadIdx.x; x < X; x += blockDim.x) dst[x] = f2bf(src[x]);
}

// W1 (4096 x 1536) fp32 -> Wd (4096 x 512) bf16 + Wh (4096 x 1024) bf16
__global__ void k_conv_w1(const float* __restrict__ W1,
                          unsigned short* __restrict__ Wd,
                          unsigned short* __restrict__ Wh) {
  const int g = blockIdx.x;
  const float* src = W1 + (size_t)g * (E_ + H_);
  for (int c = threadIdx.x; c < E_ + H_; c += blockDim.x) {
    unsigned short v = f2bf(src[c]);
    if (c < E_) Wd[(size_t)g * E_ + c] = v;
    else        Wh[(size_t)g * H_ + (c - E_)] = v;
  }
}

__global__ void k_conv_flat(const float* __restrict__ in,
                            unsigned short* __restrict__ out, int n) {
  int i = blockIdx.x * blockDim.x + threadIdx.x;
  if (i < n) out[i] = f2bf(in[i]);
}

// counters = 0; hbuf[0] = bf16(hidden[:, :H])
__global__ void k_init(const float* __restrict__ hidden,
                       unsigned short* __restrict__ hbuf,
                       int* __restrict__ counters) {
  int i = blockIdx.x * blockDim.x + threadIdx.x;   // 65536 threads
  if (i < T_) counters[i] = 0;
  int b = i >> 10, j = i & (H_ - 1);
  hbuf[(size_t)b * H_ + j] = f2bf(hidden[(size_t)b * G2 + j]);
}

// ---------------------------------------------------------------------------
// Generic bf16 WMMA GEMM:  C[M,N] = A[M,K] * B[N,K]^T + bias[N]   (fp32 out)
// Block = 256 threads (8 wave32), 64x64 output tile.
// Wave w: M-subtile = w/2, shares its A fragment across two N-subtiles.
// ---------------------------------------------------------------------------
__global__ void __launch_bounds__(256)
k_wmma_gemm_bias(const unsigned short* __restrict__ A,
                 const unsigned short* __restrict__ Bm,
                 const float* __restrict__ bias,
                 float* __restrict__ C, int N, int K) {
  const int lane = threadIdx.x & 31;
  const int wave = threadIdx.x >> 5;
  const int m0 = blockIdx.y * 64 + (wave >> 1) * 16;
  const int n0 = blockIdx.x * 64 + (wave & 1) * 32;

  v8f acc0 = {0.f,0.f,0.f,0.f,0.f,0.f,0.f,0.f};
  v8f acc1 = {0.f,0.f,0.f,0.f,0.f,0.f,0.f,0.f};
  for (int k = 0; k < K; k += 32) {
    v16bf a  = load_frag(A,  K, m0,      k, lane);
    v16bf b0 = load_frag(Bm, K, n0,      k, lane);
    v16bf b1 = load_frag(Bm, K, n0 + 16, k, lane);
    acc0 = wmma_bf16(a, b0, acc0);
    acc1 = wmma_bf16(a, b1, acc1);
  }

  const int nc0 = n0 + (lane & 15);
  const int nc1 = nc0 + 16;
  const int half = lane >> 4;
  const float bs0 = bias[nc0], bs1 = bias[nc1];
#pragma unroll
  for (int r = 0; r < 8; ++r) {
    const size_t m = (size_t)(m0 + r + 8 * half);
    C[m * N + nc0] = acc0[r] + bs0;
    C[m * N + nc1] = acc1[r] + bs1;
  }
}

// field[t*B+b, j] = sigmoid(gz[row, j]) * tanh(gz[row, H+j])
__global__ void k_field(const float* __restrict__ gz, float* __restrict__ field) {
  size_t i = (size_t)blockIdx.x * blockDim.x + threadIdx.x;  // over TB*H
  size_t row = i >> 10;
  int j = (int)(i & (H_ - 1));
  float l = gz[row * G2 + j];
  float z = gz[row * G2 + H_ + j];
  field[i] = sigmoidf_(l) * tanhf(z);
}

// ---------------------------------------------------------------------------
// Persistent recurrent kernel. 64 blocks x 256 threads; block owns hidden
// units [j0, j0+16). LDS (272 KB of the WGP's 320 KB):
//   [0      , 128K) Wh slice   64 x 1024 bf16  (TDM-loaded once, 512-step reuse)
//   [128K   , 256K) h tile     64 x 1024 bf16  (TDM-loaded every step)
//   [256K   , 272K) gate staging 64 x 64 fp32
// c lives in registers. h double-buffered bf16 in global; grid barrier
// (release fence + atomic + acquire spin w/ s_sleep) per step.
// ---------------------------------------------------------------------------
#define WHS_OFF 0u
#define HLS_OFF (64u * H_ * 2u)              // 131072
#define GSM_OFF (2u * 64u * H_ * 2u)         // 262144
#define SMEM_BYTES (GSM_OFF + 64u * 64u * 4u)

__global__ void __launch_bounds__(256, 1)
k_recurrent(const unsigned short* __restrict__ Whbf,   // 4096 x 1024 bf16
            const float* __restrict__ gates_d,         // TB x 4096
            const float* __restrict__ field,           // TB x 1024
            const float* __restrict__ hidden,          // B x 2048
            unsigned short* __restrict__ hbuf,         // 2 x B x 1024 bf16
            int* __restrict__ counters,                // T
            float* __restrict__ out,                   // B x T x 1024
            float* __restrict__ hf, float* __restrict__ cf) {
  extern __shared__ uint4 smem4[];
  unsigned short* whs = (unsigned short*)((char*)smem4 + WHS_OFF);
  unsigned short* hls = (unsigned short*)((char*)smem4 + HLS_OFF);
  float*          gsm = (float*)((char*)smem4 + GSM_OFF);

  const int tid = threadIdx.x;
  const int lane = tid & 31, wave = tid >> 5;
  const int j0 = blockIdx.x * 16;

  // ---- Wh slice preload: rows {gt*1024 + j0 .. +15} for gt=0..3 ----
#if HAVE_TDM
  if (wave == 0) {
    for (int g = 0; g < 4; ++g)
      tdm_load_2d_bf16(WHS_OFF + (unsigned)g * 16u * H_ * 2u,
                       Whbf + ((size_t)g * H_ + j0) * H_,
                       /*rows=*/16u, /*cols=*/H_, /*stride=*/H_);
  }
#else
  for (int lr = wave; lr < 64; lr += 8) {
    const int grow = (lr >> 4) * H_ + j0 + (lr & 15);
    const uint4* src = (const uint4*)(Whbf + (size_t)grow * H_);
    uint4* dst = (uint4*)(whs + (size_t)lr * H_);
    for (int c = lane; c < H_ / 8; c += 32) dst[c] = src[c];
  }
#endif

  // c state in registers: thread handles elements e0..e0+3 of the 64x16 chunk
  const int e0 = tid * 4;
  float creg[4];
#pragma unroll
  for (int q = 0; q < 4; ++q) {
    const int b = (e0 + q) >> 4, jj = (e0 + q) & 15;
    creg[q] = hidden[(size_t)b * G2 + H_ + j0 + jj];
  }
#if HAVE_TDM
  if (wave == 0) __builtin_amdgcn_s_wait_tensorcnt(0);
#endif
  __syncthreads();

  // wave -> gate-type gt = wave/2 (B fragment shared), M-subtiles {0,1}/{2,3}
  const int gt  = wave >> 1;
  const int mt0 = (wave & 1) * 2;

  for (int t = 0; t < T_; ++t) {
    const unsigned short* hb = hbuf + (size_t)(t & 1) * (B_ * H_);

#if HAVE_TDM
    // DMA this step's h (64x1024 bf16, 128 KB) into LDS via the TDM.
    if (wave == 0) {
      tdm_load_2d_bf16(HLS_OFF, hb, /*rows=*/B_, /*cols=*/H_, /*stride=*/H_);
      __builtin_amdgcn_s_wait_tensorcnt(0);
    }
    __syncthreads();
    const unsigned short* hsrc = hls;
#else
    const unsigned short* hsrc = hb;
#endif

    v8f acc0 = {0.f,0.f,0.f,0.f,0.f,0.f,0.f,0.f};
    v8f acc1 = {0.f,0.f,0.f,0.f,0.f,0.f,0.f,0.f};
#pragma unroll 4
    for (int k = 0; k < H_; k += 32) {
      v16bf bfr = load_frag(whs,  H_, gt * 16, k, lane);        // LDS
      v16bf a0  = load_frag(hsrc, H_, mt0 * 16,      k, lane);  // LDS (TDM)
      v16bf a1  = load_frag(hsrc, H_, mt0 * 16 + 16, k, lane);
      acc0 = wmma_bf16(a0, bfr, acc0);
      acc1 = wmma_bf16(a1, bfr, acc1);
    }

    // Stage gate tiles: gsm[m*64 + gt*16 + n]
    {
      const int n = lane & 15, half = lane >> 4;
#pragma unroll
      for (int r = 0; r < 8; ++r) {
        const int m0 = mt0 * 16 + r + 8 * half;
        gsm[m0 * 64 + gt * 16 + n] = acc0[r];
        gsm[(m0 + 16) * 64 + gt * 16 + n] = acc1[r];
      }
    }
    __syncthreads();

    // Elementwise LSTM cell update
    const size_t row0 = (size_t)t * B_;
#pragma unroll
    for (int q = 0; q < 4; ++q) {
      const int e = e0 + q, b = e >> 4, jj = e & 15;
      const size_t gdrow = (row0 + b) * G4 + j0 + jj;
      const float gi = gsm[b * 64 + 0 * 16 + jj] + gates_d[gdrow];
      const float gf = gsm[b * 64 + 1 * 16 + jj] + gates_d[gdrow + H_];
      const float gg = gsm[b * 64 + 2 * 16 + jj] + gates_d[gdrow + 2 * H_];
      const float go = gsm[b * 64 + 3 * 16 + jj] + gates_d[gdrow + 3 * H_];
      const float fld = field[(row0 + b) * H_ + j0 + jj];
      if (t + 1 < T_)  // prefetch next step's pre-gates into cache
        __builtin_prefetch(&gates_d[gdrow + (size_t)B_ * G4], 0, 1);

      const float cn = sigmoidf_(gf + 1.0f) * creg[q]
                     + sigmoidf_(gi) * tanhf(gg) + fld;
      const float hn = sigmoidf_(go) * tanhf(cn);
      creg[q] = cn;

      out[((size_t)b * T_ + t) * H_ + j0 + jj] = hn;
      hbuf[(size_t)((t + 1) & 1) * (B_ * H_) + (size_t)b * H_ + j0 + jj] = f2bf(hn);
      if (t == T_ - 1) {
        hf[(size_t)b * H_ + j0 + jj] = hn;
        cf[(size_t)b * H_ + j0 + jj] = cn;
      }
    }

    if (t + 1 < T_) {
      __threadfence();          // release h writes device-wide
      __syncthreads();
      if (tid == 0) {
        atomicAdd(&counters[t], 1);
        while (__hip_atomic_load(&counters[t], __ATOMIC_ACQUIRE,
                                 __HIP_MEMORY_SCOPE_AGENT) < (int)gridDim.x) {
          __builtin_amdgcn_s_sleep(1);
        }
      }
      __syncthreads();
      __threadfence();          // acquire before reading other blocks' h
    }
  }
}

// ---------------------------------------------------------------------------
// Host launcher
// ---------------------------------------------------------------------------
extern "C" void kernel_launch(void* const* d_in, const int* in_sizes, int n_in,
                              void* d_out, int out_size, void* d_ws, size_t ws_size,
                              hipStream_t stream) {
  (void)in_sizes; (void)n_in; (void)out_size; (void)ws_size;
  const float* input_d = (const float*)d_in[0];  // (B,T,E)
  const float* input_z = (const float*)d_in[1];  // (B,T,Z)
  const float* hidden  = (const float*)d_in[2];  // (B,2H)
  const float* W1      = (const float*)d_in[3];  // (4H, E+H)
  const float* b1      = (const float*)d_in[4];  // (4H,)
  const float* Wz      = (const float*)d_in[5];  // (2H, Z)
  const float* bz      = (const float*)d_in[6];  // (2H,)

  float* out = (float*)d_out;                    // (B,T,H) ++ h_f ++ c_f
  float* hf  = out + (size_t)B_ * T_ * H_;
  float* cf  = hf + (size_t)B_ * H_;

  // Workspace layout (gates_z region is reused by gates_d after k_field)
  char* ws = (char*)d_ws;
  float*          gates = (float*)(ws);                               // 512 MB
  float*          field = (float*)(ws + 0x20000000ull);               // 128 MB
  unsigned short* Ad    = (unsigned short*)(ws + 0x28000000ull);      //  32 MB
  unsigned short* Az    = (unsigned short*)(ws + 0x2A000000ull);      //   8 MB
  unsigned short* Wdbf  = (unsigned short*)(ws + 0x2A800000ull);      //   4 MB
  unsigned short* Whbf  = (unsigned short*)(ws + 0x2AC00000ull);      //   8 MB
  unsigned short* Wzbf  = (unsigned short*)(ws + 0x2B400000ull);      // 0.5 MB
  unsigned short* hbuf  = (unsigned short*)(ws + 0x2B480000ull);      // 256 KB
  int*            cnt   = (int*)(ws + 0x2B4C0000ull);                 //   2 KB

  // 1) init + precision conversion
  k_init<<<256, 256, 0, stream>>>(hidden, hbuf, cnt);
  k_conv_w1<<<G4, 256, 0, stream>>>(W1, Wdbf, Whbf);
  k_conv_flat<<<(G2 * Z_ + 255) / 256, 256, 0, stream>>>(Wz, Wzbf, G2 * Z_);
  k_conv_transpose<<<dim3(T_, B_), 128, 0, stream>>>(input_d, Ad, E_);
  k_conv_transpose<<<dim3(T_, B_), 128, 0, stream>>>(input_z, Az, Z_);

  // 2) gates_z GEMM -> field (gates buffer freed for reuse afterwards)
  k_wmma_gemm_bias<<<dim3(G2 / 64, TB / 64), 256, 0, stream>>>(Az, Wzbf, bz,
                                                               gates, G2, Z_);
  k_field<<<(int)(((size_t)TB * H_) / 256), 256, 0, stream>>>(gates, field);

  // 3) gates_d GEMM (overwrites gates_z region; field already extracted)
  k_wmma_gemm_bias<<<dim3(G4 / 64, TB / 64), 256, 0, stream>>>(Ad, Wdbf, b1,
                                                               gates, G4, E_);

  // 4) persistent recurrence: 64 WGs, 256 thr, 272 KB dynamic LDS
  k_recurrent<<<64, 256, SMEM_BYTES, stream>>>(Whbf, gates, field, hidden,
                                               hbuf, cnt, out, hf, cf);
}
